// PredictiveMaintenanceGNN_17463337025660
// MI455X (gfx1250) — compile-verified, matrix operand order checked by reading, not run on hardware
//
#include <hip/hip_runtime.h>
#include <hip/hip_bf16.h>

typedef __attribute__((ext_vector_type(16))) _Float16 v16h;
typedef __attribute__((ext_vector_type(8)))  float    v8f;

#define FDIM 128
#define LPAD 136   // padded LDS row stride in halves: 68 dwords == 4 (mod 64 banks)
#define NUM_GRAPHS 512

// ---------------------------------------------------------------------------
// init: deg = 1.0 (self loop), zero pooling accumulators
// ---------------------------------------------------------------------------
__global__ void init_kernel(float* __restrict__ deg, float* __restrict__ gsum,
                            float* __restrict__ gcnt, int N) {
  int i = blockIdx.x * blockDim.x + threadIdx.x;
  if (i < N) deg[i] = 1.0f;
  if (i < NUM_GRAPHS) { gsum[i] = 0.0f; gcnt[i] = 0.0f; }
}

// deg[dst] += 1 per edge
__global__ void count_deg_kernel(const int* __restrict__ dst, float* __restrict__ deg, int E) {
  int e = blockIdx.x * blockDim.x + threadIdx.x;
  if (e < E) atomicAdd(&deg[dst[e]], 1.0f);
}

// deg -> rsqrt(deg) in place (deg >= 1 always)
__global__ void rsqrt_kernel(float* __restrict__ deg, int N) {
  int i = blockIdx.x * blockDim.x + threadIdx.x;
  if (i < N) deg[i] = rsqrtf(deg[i]);
}

// ---------------------------------------------------------------------------
// Fragment load: two contiguous 16B runs -> 2x ds_load_b128.
// Per-lane 16-bit A/B layout (ISA 7.12.2): lanes 0-15 -> M/N=lane, K base 0;
// lanes 16-31 -> M/N=lane-16, K base 8; fragment elements 8..15 are K+16.
// ---------------------------------------------------------------------------
__device__ __forceinline__ v16h load_frag(const _Float16* __restrict__ p, int k0) {
  v16h f;
#pragma unroll
  for (int j = 0; j < 8; ++j) {
    f[j]     = p[k0 + j];
    f[j + 8] = p[k0 + 16 + j];
  }
  return f;
}

// ---------------------------------------------------------------------------
// GEMM:  HS[m, :] = (X[m, :] @ W) * dinv[m]   (also writes ACC = HS, the
// self-loop initialization of the aggregation accumulator)
// One block = 16 output rows. 8 waves; wave w owns 16x16 column tile w.
// LDS: X strip row-major f16 (padded), W TRANSPOSED f16 (padded) so both A and
// B fragments are contiguous -> ds_load_b128, conflict-free banks.
// WMMA f32_16x16x32_f16, K = 128 in 4 software-pipelined steps.
// ---------------------------------------------------------------------------
__global__ void __launch_bounds__(256)
gemm_scale_wmma(const float* __restrict__ X, const float* __restrict__ W,
                const float* __restrict__ dinv, float* __restrict__ HS,
                float* __restrict__ ACC) {
  __shared__ __align__(16) _Float16 Wt[FDIM * LPAD];   // Wt[n*LPAD + k] = W[k][n], 34 KB
  __shared__ __align__(16) _Float16 Xl[16 * LPAD];     // 4.25 KB

  const int tid   = threadIdx.x;
  const int lane  = tid & 31;
  const int ntile = tid >> 5;            // 0..7 column tile
  const int mtile = blockIdx.x;          // 16-row strip

  // stage W transposed: coalesced float4 global loads, scatter u16 LDS stores
  for (int idx = tid * 4; idx < FDIM * FDIM; idx += 256 * 4) {
    float4 w4 = *(const float4*)(W + idx);
    int k = idx >> 7;          // row of W
    int n = idx & 127;         // col of W
    Wt[(n + 0) * LPAD + k] = (_Float16)w4.x;
    Wt[(n + 1) * LPAD + k] = (_Float16)w4.y;
    Wt[(n + 2) * LPAD + k] = (_Float16)w4.z;
    Wt[(n + 3) * LPAD + k] = (_Float16)w4.w;
  }
  // stage the 16x128 A strip (shared by all 8 waves): 2048 f32, 8 per thread
  {
    const float* Xt = X + (size_t)mtile * 16 * FDIM;
    int idx = tid * 8;
    int row = idx >> 7, col = idx & 127;
    float4 a0 = *(const float4*)(Xt + idx);
    float4 a1 = *(const float4*)(Xt + idx + 4);
    _Float16* xd = Xl + row * LPAD + col;
    xd[0] = (_Float16)a0.x; xd[1] = (_Float16)a0.y;
    xd[2] = (_Float16)a0.z; xd[3] = (_Float16)a0.w;
    xd[4] = (_Float16)a1.x; xd[5] = (_Float16)a1.y;
    xd[6] = (_Float16)a1.z; xd[7] = (_Float16)a1.w;
  }
  __syncthreads();

  const int mn = lane & 15;
  const int kb = (lane < 16) ? 0 : 8;
  const _Float16* arow = Xl + mn * LPAD + kb;                 // A: row mn
  const _Float16* bcol = Wt + (ntile * 16 + mn) * LPAD + kb;  // B: col n

  // software-pipelined: fragments for step i+1 load behind step i's WMMA
  v8f acc = {};
  v16h a0 = load_frag(arow, 0),  b0 = load_frag(bcol, 0);
  v16h a1 = load_frag(arow, 32), b1 = load_frag(bcol, 32);
  acc = __builtin_amdgcn_wmma_f32_16x16x32_f16(false, a0, false, b0, (short)0, acc, false, false);
  a0 = load_frag(arow, 64); b0 = load_frag(bcol, 64);
  acc = __builtin_amdgcn_wmma_f32_16x16x32_f16(false, a1, false, b1, (short)0, acc, false, false);
  a1 = load_frag(arow, 96); b1 = load_frag(bcol, 96);
  acc = __builtin_amdgcn_wmma_f32_16x16x32_f16(false, a0, false, b0, (short)0, acc, false, false);
  acc = __builtin_amdgcn_wmma_f32_16x16x32_f16(false, a1, false, b1, (short)0, acc, false, false);

  // C layout: VGPR r -> row (lane<16 ? r : r+8), col = ntile*16 + (lane&15)
  const int colg  = ntile * 16 + mn;
  const int rbase = mtile * 16 + ((lane < 16) ? 0 : 8);
#pragma unroll
  for (int r = 0; r < 8; ++r) {
    int row = rbase + r;
    float v = acc[r] * dinv[row];
    HS [(size_t)row * FDIM + colg] = v;
    ACC[(size_t)row * FDIM + colg] = v;   // self-loop init of accumulator
  }
}

// ---------------------------------------------------------------------------
// Edge aggregation: ACC[dst, :] += HS[src, :]   (norm already folded in)
// One wave per edge, lane handles 4 features (float4 gather, 4 f32 atomics).
// ---------------------------------------------------------------------------
__global__ void __launch_bounds__(256)
edge_agg_kernel(const float* __restrict__ HS, float* __restrict__ ACC,
                const int* __restrict__ src, const int* __restrict__ dst, int E) {
  int e = blockIdx.x * 8 + (threadIdx.x >> 5);
  if (e >= E) return;
  int lane = threadIdx.x & 31;
  __builtin_prefetch(src + e + 2048, 0, 1);   // global_prefetch_b8
  int s = src[e], d = dst[e];
  float4 v = *(const float4*)(HS + (size_t)s * FDIM + lane * 4);
  float* a = ACC + (size_t)d * FDIM + lane * 4;
  atomicAdd(a + 0, v.x); atomicAdd(a + 1, v.y);
  atomicAdd(a + 2, v.z); atomicAdd(a + 3, v.w);
}

// ---------------------------------------------------------------------------
// Epilogue: ACC = [relu](ACC * dinv[row] + bias)   in place
// ---------------------------------------------------------------------------
__global__ void __launch_bounds__(256)
epilogue_kernel(float* __restrict__ ACC, const float* __restrict__ dinv,
                const float* __restrict__ bias, int N, int do_relu) {
  int t = blockIdx.x * blockDim.x + threadIdx.x;    // one float4 each
  int total = N * (FDIM / 4);
  if (t >= total) return;
  int row = t / (FDIM / 4);
  int c4  = (t % (FDIM / 4)) * 4;
  float di = dinv[row];
  float4 v = *(float4*)(ACC + (size_t)row * FDIM + c4);
  float4 b = *(const float4*)(bias + c4);
  v.x = v.x * di + b.x; v.y = v.y * di + b.y;
  v.z = v.z * di + b.z; v.w = v.w * di + b.w;
  if (do_relu) {
    v.x = fmaxf(v.x, 0.0f); v.y = fmaxf(v.y, 0.0f);
    v.z = fmaxf(v.z, 0.0f); v.w = fmaxf(v.w, 0.0f);
  }
  *(float4*)(ACC + (size_t)row * FDIM + c4) = v;
}

// ---------------------------------------------------------------------------
// Fused pool + linear head (O==1): per node p = dot(h2[node,:], Wl);
// gsum[batch[node]] += p; gcnt[batch[node]] += 1.  One wave per node.
// ---------------------------------------------------------------------------
__global__ void __launch_bounds__(256)
pool_kernel(const float* __restrict__ H2, const float* __restrict__ Wlin,
            const int* __restrict__ batch, float* __restrict__ gsum,
            float* __restrict__ gcnt, int N) {
  int node = blockIdx.x * 8 + (threadIdx.x >> 5);
  if (node >= N) return;
  int lane = threadIdx.x & 31;
  float4 h = *(const float4*)(H2 + (size_t)node * FDIM + lane * 4);
  float4 w = *(const float4*)(Wlin + lane * 4);
  float v = h.x * w.x + h.y * w.y + h.z * w.z + h.w * w.w;
#pragma unroll
  for (int off = 16; off > 0; off >>= 1) v += __shfl_down(v, off, 32);
  if (lane == 0) {
    int g = batch[node];
    atomicAdd(&gsum[g], v);
    atomicAdd(&gcnt[g], 1.0f);
  }
}

__global__ void final_kernel(const float* __restrict__ gsum, const float* __restrict__ gcnt,
                             const float* __restrict__ bl, float* __restrict__ out) {
  int g = blockIdx.x * blockDim.x + threadIdx.x;
  if (g < NUM_GRAPHS) out[g] = gsum[g] / fmaxf(gcnt[g], 1.0f) + bl[0];
}

// ---------------------------------------------------------------------------
extern "C" void kernel_launch(void* const* d_in, const int* in_sizes, int n_in,
                              void* d_out, int out_size, void* d_ws, size_t ws_size,
                              hipStream_t stream) {
  const float* x    = (const float*)d_in[0];
  const int*   ei   = (const int*)  d_in[1];
  // d_in[2] = edge_attr: unused by the reference model
  const int*   batch= (const int*)  d_in[3];
  const float* W1   = (const float*)d_in[4];
  const float* b1   = (const float*)d_in[5];
  const float* W2   = (const float*)d_in[6];
  const float* b2   = (const float*)d_in[7];
  const float* Wlin = (const float*)d_in[8];
  const float* bl   = (const float*)d_in[9];
  float* out = (float*)d_out;

  const int N = in_sizes[0] / FDIM;   // 100000 (multiple of 16)
  const int E = in_sizes[1] / 2;      // 1600000
  const int* src = ei;
  const int* dst = ei + E;

  // workspace layout (floats): dinv[N] | hs[N*128] | acc[N*128] | gsum[512] | gcnt[512]
  float* dinv = (float*)d_ws;
  float* hs   = dinv + N;
  float* acc  = hs + (size_t)N * FDIM;
  float* gsum = acc + (size_t)N * FDIM;
  float* gcnt = gsum + NUM_GRAPHS;

  const int rowBlocks  = N / 16;            // 6250
  const int edgeBlocks = (E + 7) / 8;
  const int eltBlocks  = (N * (FDIM / 4) + 255) / 256;

  // degree / norm (shared by both layers) + zero pooling accumulators
  init_kernel     <<<(N + 255) / 256, 256, 0, stream>>>(dinv, gsum, gcnt, N);
  count_deg_kernel<<<(E + 255) / 256, 256, 0, stream>>>(dst, dinv, E);
  rsqrt_kernel    <<<(N + 255) / 256, 256, 0, stream>>>(dinv, N);

  // ---- layer 1: h1 = relu(gcn(x, W1, b1)) ----
  gemm_scale_wmma <<<rowBlocks, 256, 0, stream>>>(x, W1, dinv, hs, acc);
  edge_agg_kernel <<<edgeBlocks, 256, 0, stream>>>(hs, acc, src, dst, E);
  epilogue_kernel <<<eltBlocks, 256, 0, stream>>>(acc, dinv, b1, N, 1);

  // ---- layer 2: h2 = gcn(h1, W2, b2)  (h1 lives in acc; safe in-place:
  // each GEMM block stages its own 16 rows to LDS before overwriting them) ----
  gemm_scale_wmma <<<rowBlocks, 256, 0, stream>>>(acc, W2, dinv, hs, acc);
  edge_agg_kernel <<<edgeBlocks, 256, 0, stream>>>(hs, acc, src, dst, E);
  epilogue_kernel <<<eltBlocks, 256, 0, stream>>>(acc, dinv, b2, N, 0);

  // ---- pool (mean per graph) fused with 128->1 linear head ----
  pool_kernel <<<(N + 7) / 8, 256, 0, stream>>>(acc, Wlin, batch, gsum, gcnt, N);
  final_kernel<<<(NUM_GRAPHS + 255) / 256, 256, 0, stream>>>(gsum, gcnt, bl, out);
}
